// MyModel_19662360281110
// MI455X (gfx1250) — matrix operand order repaired and checked by reading, not compiled, not run on hardware
//
#include <hip/hip_runtime.h>
#include <hip/hip_bf16.h>

typedef __attribute__((ext_vector_type(16))) __bf16 v16bf;
typedef __attribute__((ext_vector_type(2)))  __bf16 v2bf;
typedef __attribute__((ext_vector_type(8)))  float  v8f;
typedef int v4i_g __attribute__((vector_size(16)));

#define B_SZ   8192
#define D_INP  8192
#define U0     1024
#define U1     256
#define NHEAD  6
#define RAW_INIT_STD 4.99323934837f   // log(exp(5)-1)
#define MIN_STD 1e-4f
#define MEAN_SCALE 5.0f

#define GLOBAL_AS __attribute__((address_space(1)))
#define LDS_AS    __attribute__((address_space(3)))

// ---- fp32 -> bf16 round-to-nearest-even, packed pair ----
__device__ __forceinline__ unsigned pk2bf(float lo, float hi) {
#if __has_builtin(__builtin_amdgcn_cvt_pk_bf16_f32)
    v2bf r = __builtin_amdgcn_cvt_pk_bf16_f32(lo, hi);
    return __builtin_bit_cast(unsigned, r);
#else
    unsigned r;
    asm("v_cvt_pk_bf16_f32 %0, %1, %2" : "=v"(r) : "v"(lo), "v"(hi));
    return r;
#endif
}

#if __has_builtin(__builtin_amdgcn_global_load_async_to_lds_b128)
#define USE_ASYNC_LDS 1
__device__ __forceinline__ void async_cp16(const unsigned short* g, unsigned short* l) {
    __builtin_amdgcn_global_load_async_to_lds_b128(
        (GLOBAL_AS v4i_g*)g, (LDS_AS v4i_g*)l, 0, 0);
}
__device__ __forceinline__ void wait_async0() {
#if __has_builtin(__builtin_amdgcn_s_wait_asynccnt)
    __builtin_amdgcn_s_wait_asynccnt(0);
#else
    asm volatile("s_wait_asynccnt 0x0" ::: "memory");
#endif
}
#else
__device__ __forceinline__ void async_cp16(const unsigned short* g, unsigned short* l) {
    *(uint4*)l = *(const uint4*)g;          // sync fallback
}
__device__ __forceinline__ void wait_async0() {}
#endif

union Frag { uint4 q[2]; v16bf v; };

// =====================================================================
// Kernel 0: convert+transpose fp32 [batch][R][C] -> bf16 [batch][C][R]
// 64x64 tiles via LDS.
// =====================================================================
__global__ __launch_bounds__(256)
void cvt_transpose(const float* __restrict__ in, unsigned short* __restrict__ outT,
                   int R, int C) {
    __shared__ __align__(8) unsigned short T[64][72];
    const int tid = threadIdx.x;
    const float* inb = in + (size_t)blockIdx.z * R * C;
    unsigned short* outb = outT + (size_t)blockIdx.z * R * C;
    const int r0 = blockIdx.y * 64, c0 = blockIdx.x * 64;
#pragma unroll
    for (int i = 0; i < 4; ++i) {
        const int idx = tid + i * 256;        // 1024 float4
        const int rr = idx >> 4;              // 16 float4 per row
        const int cc = (idx & 15) << 2;
        const float4 v = *(const float4*)(inb + (size_t)(r0 + rr) * C + c0 + cc);
        *(unsigned*)&T[rr][cc]     = pk2bf(v.x, v.y);
        *(unsigned*)&T[rr][cc + 2] = pk2bf(v.z, v.w);
    }
    __syncthreads();
#pragma unroll
    for (int i = 0; i < 4; ++i) {
        const int idx = tid + i * 256;        // 1024 uint2 (4 bf16 each)
        const int cc = idx >> 4;
        const int rr = (idx & 15) << 2;
        const unsigned e0 = T[rr + 0][cc], e1 = T[rr + 1][cc];
        const unsigned e2 = T[rr + 2][cc], e3 = T[rr + 3][cc];
        uint2 o; o.x = e0 | (e1 << 16); o.y = e2 | (e3 << 16);
        *(uint2*)(outb + (size_t)(c0 + cc) * R + r0 + rr) = o;
    }
}

// =====================================================================
// Kernel 1: H = relu(X @ Wfc + bfc) -> bf16 [8192,1024]
// 128(M)x256(N) tile/WG, 8 waves of 64x64, double-buffered LDS.
// A: fp32 load + v_cvt_pk_bf16_f32 + ds_store.  B: async copy from WfcT.
// =====================================================================
#define NK1 (D_INP / 32)
__global__ __launch_bounds__(256)
void trunk_gemm(const float* __restrict__ X, const unsigned short* __restrict__ WfcT,
                const float* __restrict__ bfc, unsigned short* __restrict__ H) {
    __shared__ __align__(16) unsigned short As[2][128][40];  // [M][K]
    __shared__ __align__(16) unsigned short Bs[2][256][40];  // [N][K]

    const int tid  = threadIdx.x;
    const int lane = tid & 31;
    const int wave = tid >> 5;
    const int wm   = wave >> 2;            // 0..1
    const int wn   = wave & 3;             // 0..3
    const int mBase = blockIdx.y * 128;
    const int nBase = blockIdx.x * 256;

    v8f acc[4][4];
#pragma unroll
    for (int m = 0; m < 4; ++m)
#pragma unroll
        for (int n = 0; n < 4; ++n) acc[m][n] = (v8f){0,0,0,0,0,0,0,0};

    float4 aReg[4];
    auto loadA = [&](int k0) {
#pragma unroll
        for (int i = 0; i < 4; ++i) {
            const int idx = tid + i * 256;
            const int r = idx >> 3;
            const int c = (idx & 7) << 2;
            const float* gp = X + (size_t)(mBase + r) * D_INP + k0 + c;
            aReg[i] = *(const float4*)gp;
            if (k0 + 32 < D_INP) __builtin_prefetch(gp + 32, 0, 1);
        }
    };
    auto storeA = [&](int buf) {
#pragma unroll
        for (int i = 0; i < 4; ++i) {
            const int idx = tid + i * 256;
            const int r = idx >> 3;
            const int c = (idx & 7) << 2;
            uint2 p; p.x = pk2bf(aReg[i].x, aReg[i].y); p.y = pk2bf(aReg[i].z, aReg[i].w);
            *(uint2*)&As[buf][r][c] = p;
        }
    };
    auto loadB = [&](int k0, int buf) {
#pragma unroll
        for (int i = 0; i < 4; ++i) {
            const int idx = tid + i * 256;    // 1024 x b128 chunks
            const int n = idx >> 2;
            const int c = (idx & 3) << 3;
            async_cp16(WfcT + (size_t)(nBase + n) * D_INP + k0 + c, &Bs[buf][n][c]);
        }
    };
    auto compute = [&](int buf) {
        Frag a[4], b[4];
#pragma unroll
        for (int m = 0; m < 4; ++m) {
            const unsigned short* p = &As[buf][wm * 64 + m * 16 + (lane & 15)][(lane >> 4) * 8];
            a[m].q[0] = *(const uint4*)p;
            a[m].q[1] = *(const uint4*)(p + 16);
        }
#pragma unroll
        for (int n = 0; n < 4; ++n) {
            const unsigned short* p = &Bs[buf][wn * 64 + n * 16 + (lane & 15)][(lane >> 4) * 8];
            b[n].q[0] = *(const uint4*)p;
            b[n].q[1] = *(const uint4*)(p + 16);
        }
#pragma unroll
        for (int m = 0; m < 4; ++m)
#pragma unroll
            for (int n = 0; n < 4; ++n)
                acc[m][n] = __builtin_amdgcn_wmma_f32_16x16x32_bf16(
                    false, a[m].v, false, b[n].v, (short)0, acc[m][n], false, false);
    };

    loadA(0);
    loadB(0, 0);
    storeA(0);
    wait_async0();
    __syncthreads();
#pragma unroll 1
    for (int kt = 0; kt < NK1; ++kt) {
        const int cur = kt & 1, nxt = cur ^ 1;
        const bool more = (kt + 1 < NK1);
        if (more) { loadA((kt + 1) * 32); loadB((kt + 1) * 32, nxt); }
        compute(cur);
        if (more) { storeA(nxt); wait_async0(); }
        __syncthreads();
    }

    // Epilogue: bias + relu, packed bf16 stores
#pragma unroll
    for (int m = 0; m < 4; ++m) {
#pragma unroll
        for (int n = 0; n < 4; ++n) {
            const int C = nBase + wn * 64 + n * 16 + (lane & 15);
            const float bias = bfc[C];
#pragma unroll
            for (int i = 0; i < 8; i += 2) {
                const int R0 = mBase + wm * 64 + m * 16 + i + ((lane >> 4) << 3);
                const unsigned pk = pk2bf(fmaxf(acc[m][n][i] + bias, 0.0f),
                                          fmaxf(acc[m][n][i + 1] + bias, 0.0f));
                H[(size_t)R0 * U0 + C]       = (unsigned short)pk;
                H[(size_t)(R0 + 1) * U0 + C] = (unsigned short)(pk >> 16);
            }
        }
    }
}

// =====================================================================
// Kernel 2: per head h: Z = relu(H @ W1[h] + b1[h]); store rows with
// command[row]==h into Zsel fp32 [8192,256].  Fully async-staged,
// double-buffered: hot loop = async copies + ds frag loads + WMMA.
// =====================================================================
#define NK2 (U0 / 32)
__global__ __launch_bounds__(256)
void heads_gemm(const unsigned short* __restrict__ H, const unsigned short* __restrict__ W1T,
                const float* __restrict__ b1, const int* __restrict__ cmd,
                float* __restrict__ Zsel) {
    __shared__ __align__(16) unsigned short As[2][128][40];
    __shared__ __align__(16) unsigned short Bs[2][128][40];

    const int tid  = threadIdx.x;
    const int lane = tid & 31;
    const int wave = tid >> 5;
    const int wm   = wave >> 2;
    const int wn   = wave & 3;
    const int head  = blockIdx.z;
    const int mBase = blockIdx.y * 128;
    const int nBase = blockIdx.x * 128;
    const unsigned short* W1h = W1T + (size_t)head * U0 * U1;   // [N=256][K=1024] bf16

    v8f acc[4][2];
#pragma unroll
    for (int m = 0; m < 4; ++m)
#pragma unroll
        for (int n = 0; n < 2; ++n) acc[m][n] = (v8f){0,0,0,0,0,0,0,0};

    auto stage = [&](int k0, int buf) {
#pragma unroll
        for (int i = 0; i < 2; ++i) {        // A: 512 x b128 chunks
            const int idx = tid + i * 256;
            const int r = idx >> 2;
            const int c = (idx & 3) << 3;
            async_cp16(H + (size_t)(mBase + r) * U0 + k0 + c, &As[buf][r][c]);
        }
#pragma unroll
        for (int i = 0; i < 2; ++i) {        // B: 512 x b128 chunks
            const int idx = tid + i * 256;
            const int n = idx >> 2;
            const int c = (idx & 3) << 3;
            async_cp16(W1h + (size_t)(nBase + n) * U0 + k0 + c, &Bs[buf][n][c]);
        }
    };
    auto compute = [&](int buf) {
        Frag a[4], b[2];
#pragma unroll
        for (int m = 0; m < 4; ++m) {
            const unsigned short* p = &As[buf][wm * 64 + m * 16 + (lane & 15)][(lane >> 4) * 8];
            a[m].q[0] = *(const uint4*)p;
            a[m].q[1] = *(const uint4*)(p + 16);
        }
#pragma unroll
        for (int n = 0; n < 2; ++n) {
            const unsigned short* p = &Bs[buf][wn * 32 + n * 16 + (lane & 15)][(lane >> 4) * 8];
            b[n].q[0] = *(const uint4*)p;
            b[n].q[1] = *(const uint4*)(p + 16);
        }
#pragma unroll
        for (int m = 0; m < 4; ++m)
#pragma unroll
            for (int n = 0; n < 2; ++n)
                acc[m][n] = __builtin_amdgcn_wmma_f32_16x16x32_bf16(
                    false, a[m].v, false, b[n].v, (short)0, acc[m][n], false, false);
    };

    stage(0, 0);
    wait_async0();
    __syncthreads();
#pragma unroll 1
    for (int kt = 0; kt < NK2; ++kt) {
        const int cur = kt & 1, nxt = cur ^ 1;
        const bool more = (kt + 1 < NK2);
        if (more) stage((kt + 1) * 32, nxt);
        compute(cur);
        if (more) wait_async0();
        __syncthreads();
    }

    // Epilogue: bias + relu, store fp32 only where command matches this head
#pragma unroll
    for (int m = 0; m < 4; ++m) {
        const int Rbase = mBase + wm * 64 + m * 16 + ((lane >> 4) << 3);
        int cmds[8];
#pragma unroll
        for (int i = 0; i < 8; ++i) cmds[i] = cmd[Rbase + i];
#pragma unroll
        for (int n = 0; n < 2; ++n) {
            const int C = nBase + wn * 32 + n * 16 + (lane & 15);
            const float bias = b1[head * U1 + C];
#pragma unroll
            for (int i = 0; i < 8; ++i) {
                if (cmds[i] == head) {
                    const float val = fmaxf(acc[m][n][i] + bias, 0.0f);
                    Zsel[(size_t)(Rbase + i) * U1 + C] = val;
                }
            }
        }
    }
}

// =====================================================================
// Kernel 3: y = Zsel @ W2[cmd] + b2[cmd]; tanh/softplus epilogue.
// =====================================================================
__global__ __launch_bounds__(256)
void head_out(const float* __restrict__ Z, const float* __restrict__ W2,
              const float* __restrict__ b2, const int* __restrict__ cmd,
              float* __restrict__ out) {
    const int b = blockIdx.x * 256 + threadIdx.x;
    const int h = cmd[b];
    const float* z = Z  + (size_t)b * U1;
    const float* w = W2 + (size_t)h * U1 * 4;
    float a0 = 0.f, a1 = 0.f, a2 = 0.f, a3 = 0.f;
#pragma unroll 4
    for (int i = 0; i < U1; ++i) {
        const float  zi = z[i];
        const float4 wv = *(const float4*)(w + i * 4);
        a0 = fmaf(zi, wv.x, a0);
        a1 = fmaf(zi, wv.y, a1);
        a2 = fmaf(zi, wv.z, a2);
        a3 = fmaf(zi, wv.w, a3);
    }
    a0 += b2[h * 4 + 0];
    a1 += b2[h * 4 + 1];
    a2 += b2[h * 4 + 2];
    a3 += b2[h * 4 + 3];

    const float m0 = MEAN_SCALE * tanhf(a0 * (1.0f / MEAN_SCALE));
    const float m1 = MEAN_SCALE * tanhf(a1 * (1.0f / MEAN_SCALE));
    const float x2 = a2 + RAW_INIT_STD;
    const float x3 = a3 + RAW_INIT_STD;
    const float s0 = ((x2 > 20.f) ? x2 : log1pf(expf(x2))) + MIN_STD;
    const float s1 = ((x3 > 20.f) ? x3 : log1pf(expf(x3))) + MIN_STD;

    out[b * 2 + 0] = m0;
    out[b * 2 + 1] = m1;
    out[B_SZ * 2 + b * 2 + 0] = s0;
    out[B_SZ * 2 + b * 2 + 1] = s1;
}

extern "C" void kernel_launch(void* const* d_in, const int* in_sizes, int n_in,
                              void* d_out, int out_size, void* d_ws, size_t ws_size,
                              hipStream_t stream) {
    const float* x       = (const float*)d_in[0];
    const int*   command = (const int*)  d_in[1];
    const float* W_fc    = (const float*)d_in[2];
    const float* b_fc    = (const float*)d_in[3];
    const float* W1      = (const float*)d_in[4];
    const float* b1      = (const float*)d_in[5];
    const float* W2      = (const float*)d_in[6];
    const float* b2      = (const float*)d_in[7];
    float* out = (float*)d_out;

    char* ws = (char*)d_ws;
    unsigned short* Hbf  = (unsigned short*)(ws);                               // 16 MB
    float*          Zsel = (float*)         (ws + (size_t)16 * 1024 * 1024);    //  8 MB
    unsigned short* WfcT = (unsigned short*)(ws + (size_t)24 * 1024 * 1024);    // 16 MB
    unsigned short* W1T  = (unsigned short*)(ws + (size_t)40 * 1024 * 1024);    //  3 MB

    // Weight convert+transpose (cheap, fully deterministic each call)
    cvt_transpose<<<dim3(U0 / 64, D_INP / 64, 1), 256, 0, stream>>>(W_fc, WfcT, D_INP, U0);
    cvt_transpose<<<dim3(U1 / 64, U0 / 64, NHEAD), 256, 0, stream>>>(W1, W1T, U0, U1);

    dim3 g1(U0 / 256, B_SZ / 128);            // 4 x 64  (x = N so M-panel shares L2)
    trunk_gemm<<<g1, 256, 0, stream>>>(x, WfcT, b_fc, Hbf);

    dim3 g2(U1 / 128, B_SZ / 128, NHEAD);     // 2 x 64 x 6
    heads_gemm<<<g2, 256, 0, stream>>>(Hbf, W1T, b1, command, Zsel);

    head_out<<<B_SZ / 256, 256, 0, stream>>>(Zsel, W2, b2, command, out);
}
